// MoELayer_41686952575625
// MI455X (gfx1250) — compile-verified, matrix operand order checked by reading, not run on hardware
//
#include <hip/hip_runtime.h>
#include <hip/hip_bf16.h>
#include <math.h>

#define DD   2048
#define EE   8
#define HH   5461
#define TT   4096
#define HPAD 5472           // HH rounded up to 32
#define XST  2072           // LDS row stride (elems) for X tile (4144B: conflict-free)
#define HST  5480           // LDS row stride (elems) for H tile (10960B: conflict-free)
#define TM   16             // token tile

typedef __attribute__((ext_vector_type(16))) __bf16 v16bf;
typedef __attribute__((ext_vector_type(8)))  float  v8f;

union BF16x16 {
    v16bf v;
    uint4 q[2];
    unsigned int u[8];
    unsigned short s[16];
};

__device__ __forceinline__ unsigned short f2bf(float f) {          // truncate
    union { float f; unsigned int u; } c; c.f = f;
    return (unsigned short)(c.u >> 16);
}
__device__ __forceinline__ unsigned short f2bf_rn(float f) {       // round-nearest-even
    union { float f; unsigned int u; } c; c.f = f;
    unsigned int u = c.u + 0x7FFFu + ((c.u >> 16) & 1u);
    return (unsigned short)(u >> 16);
}
__device__ __forceinline__ unsigned int pk(float a, float b) {
    return ((unsigned int)f2bf(b) << 16) | (unsigned int)f2bf(a);
}
__device__ __forceinline__ unsigned int pk_rn(float a, float b) {
    return ((unsigned int)f2bf_rn(b) << 16) | (unsigned int)f2bf_rn(a);
}

// ---------------- Weight pre-conversion: f32 -> bf16 -----------------------
__global__ __launch_bounds__(256) void conv_copy_bf16(
    const float* __restrict__ src, unsigned short* __restrict__ dst, long n4)
{
    long i = (long)blockIdx.x * blockDim.x + threadIdx.x;
    const long stride = (long)gridDim.x * blockDim.x;
    for (; i < n4; i += stride) {
        float4 f = ((const float4*)src)[i];
        ((uint2*)dst)[i] = make_uint2(pk_rn(f.x, f.y), pk_rn(f.z, f.w));
    }
}

// w2 [E][D][H] f32 -> [E][D][HPAD] bf16 with zero padding
__global__ __launch_bounds__(256) void conv_w2_pad(
    const float* __restrict__ src, unsigned short* __restrict__ dst)
{
    const long total = (long)EE * DD * HPAD;
    long i = (long)blockIdx.x * blockDim.x + threadIdx.x;
    const long stride = (long)gridDim.x * blockDim.x;
    for (; i < total; i += stride) {
        const long hp  = i % HPAD;
        const long row = i / HPAD;               // e*DD + d
        const float v = (hp < HH) ? src[row * HH + hp] : 0.0f;
        dst[i] = f2bf_rn(v);
    }
}

// ---------------- Gating: logits, softmax, top-2, lists, prob sums ----------
__global__ __launch_bounds__(256) void gate_kernel(
    const float* __restrict__ x, const float* __restrict__ wg,
    float* __restrict__ cw, float* __restrict__ probSum,
    int* __restrict__ cnt, int* __restrict__ lists)
{
    const int t    = blockIdx.x * 8 + (threadIdx.x >> 5);
    const int lane = threadIdx.x & 31;
    const float* xr = x + (size_t)t * DD;

    float acc[EE];
#pragma unroll
    for (int e = 0; e < EE; ++e) acc[e] = 0.0f;

    for (int i = 0; i < DD / 32; ++i) {
        const int d = i * 32 + lane;
        const float xv = xr[d];
#pragma unroll
        for (int e = 0; e < EE; ++e) acc[e] += xv * wg[e * DD + d];
    }
#pragma unroll
    for (int e = 0; e < EE; ++e)
        for (int off = 16; off > 0; off >>= 1)
            acc[e] += __shfl_xor(acc[e], off, 32);

    if (lane == 0) {
        float mx = acc[0];
#pragma unroll
        for (int e = 1; e < EE; ++e) mx = fmaxf(mx, acc[e]);
        float p[EE], s = 0.0f;
#pragma unroll
        for (int e = 0; e < EE; ++e) { p[e] = __expf(acc[e] - mx); s += p[e]; }
#pragma unroll
        for (int e = 0; e < EE; ++e) atomicAdd(&probSum[e], p[e] / s);

        int i0 = 0;
#pragma unroll
        for (int e = 1; e < EE; ++e) if (acc[e] > acc[i0]) i0 = e;
        int i1 = (i0 == 0) ? 1 : 0;
#pragma unroll
        for (int e = 0; e < EE; ++e) if (e != i0 && acc[e] > acc[i1]) i1 = e;

        const float e1 = __expf(acc[i1] - acc[i0]);
        const float p0 = 1.0f / (1.0f + e1);
        const float p1 = e1 / (1.0f + e1);

        float row[EE];
#pragma unroll
        for (int e = 0; e < EE; ++e) row[e] = 0.0f;
        row[i0] = p0; row[i1] = p1;
#pragma unroll
        for (int e = 0; e < EE; ++e) cw[(size_t)t * EE + e] = row[e];

        int pos0 = atomicAdd(&cnt[i0], 1); lists[i0 * TT + pos0] = t;
        int pos1 = atomicAdd(&cnt[i1], 1); lists[i1 * TT + pos1] = t;
    }
}

// ---------------- Load-balance loss ----------------------------------------
__global__ void loss_kernel(const float* __restrict__ probSum, float* __restrict__ out_loss)
{
    if (threadIdx.x == 0 && blockIdx.x == 0) {
        float mp[EE], mean = 0.0f;
#pragma unroll
        for (int e = 0; e < EE; ++e) { mp[e] = probSum[e] / (float)TT; mean += mp[e]; }
        mean /= (float)EE;
        float var = 0.0f;
#pragma unroll
        for (int e = 0; e < EE; ++e) { float d = mp[e] - mean; var += d * d; }
        var /= (float)(EE - 1);
        float lb = sqrtf(var) / (mean + 1e-10f);
        *out_loss = 0.01f * lb * lb;
    }
}

// ---------------- Main routed SwiGLU via bf16 WMMA -------------------------
template <bool PRE>
__global__ __launch_bounds__(256) void moe_main(
    const float* __restrict__ x,
    const float* __restrict__ w1f, const float* __restrict__ w2f, const float* __restrict__ w3f,
    const unsigned short* __restrict__ w1b, const unsigned short* __restrict__ w2b,
    const unsigned short* __restrict__ w3b,
    const float* __restrict__ cw, const int* __restrict__ lists, const int* __restrict__ cnt,
    float* __restrict__ out)
{
    __shared__ alignas(16) unsigned short Xs[TM * XST]; // bf16 X tile  [16, D]
    __shared__ alignas(16) unsigned short Hs[TM * HST]; // bf16 h tile  [16, HPAD] (cw-weighted)
    __shared__ int   tok_s[TM];
    __shared__ float cw_s[TM];

    const int e    = blockIdx.x >> 8;    // 256 tiles per expert
    const int tile = blockIdx.x & 255;
    const int n_e  = cnt[e];
    if (tile * TM >= n_e) return;        // uniform per block

    const int tid   = threadIdx.x;
    const int lane  = tid & 31;
    const int wave  = tid >> 5;          // 8 waves
    const int nn    = lane & 15;         // A row / B col / D col within tile
    const int khalf = lane >> 4;         // K-half select for A/B fragments
    const int off8  = khalf * 8;         // C/D row offset for upper lanes

    if (tid < TM) {
        const int r = tile * TM + tid;
        const int tok = (r < n_e) ? lists[e * TT + r] : -1;
        tok_s[tid] = tok;
        cw_s[tid]  = (tok >= 0) ? cw[(size_t)tok * EE + e] : 0.0f;
    }
    __syncthreads();

    // ---- stage X tile into LDS as bf16 (zero rows for padding tokens) ----
    for (int q = tid; q < TM * (DD / 4); q += 256) {
        const int r  = q >> 9;             // DD/4 == 512
        const int c4 = (q & 511) << 2;
        const int tok = tok_s[r];
        float4 f = make_float4(0.f, 0.f, 0.f, 0.f);
        if (tok >= 0) f = *(const float4*)&x[(size_t)tok * DD + c4];
        *(uint2*)&Xs[r * XST + c4] = make_uint2(pk(f.x, f.y), pk(f.z, f.w));
    }
    __syncthreads();

    // ---- Phase 1: h = silu(X W1^T) * (X W3^T) * cw   -> LDS (bf16) ----
    if constexpr (PRE) {
        // two h-tiles per iteration: A fragment feeds 4 WMMAs
        for (int it = wave; it < HPAD / 32; it += 8) {
            const int h0    = it * 32;
            const int hcolA = h0 + nn;
            const int hcolB = h0 + 16 + nn;
            const int hrowA = (hcolA < HH) ? hcolA : (HH - 1);
            const int hrowB = (hcolB < HH) ? hcolB : (HH - 1);
            const unsigned short* r1a = w1b + ((size_t)e * HH + hrowA) * DD;
            const unsigned short* r3a = w3b + ((size_t)e * HH + hrowA) * DD;
            const unsigned short* r1c = w1b + ((size_t)e * HH + hrowB) * DD;
            const unsigned short* r3c = w3b + ((size_t)e * HH + hrowB) * DD;

            v8f a1a = {}, a3a = {}, a1b = {}, a3b = {};
            for (int ks = 0; ks < DD; ks += 32) {
                const int c0 = ks + khalf * 8;

                BF16x16 A;
                const unsigned short* ap = &Xs[nn * XST + c0];
                A.q[0] = *(const uint4*)ap;
                A.q[1] = *(const uint4*)(ap + 16);

                BF16x16 B1a, B3a, B1b, B3b;
                B1a.q[0] = *(const uint4*)(r1a + c0);
                B1a.q[1] = *(const uint4*)(r1a + c0 + 16);
                B3a.q[0] = *(const uint4*)(r3a + c0);
                B3a.q[1] = *(const uint4*)(r3a + c0 + 16);
                B1b.q[0] = *(const uint4*)(r1c + c0);
                B1b.q[1] = *(const uint4*)(r1c + c0 + 16);
                B3b.q[0] = *(const uint4*)(r3c + c0);
                B3b.q[1] = *(const uint4*)(r3c + c0 + 16);

                a1a = __builtin_amdgcn_wmma_f32_16x16x32_bf16(false, A.v, false, B1a.v,
                                                              (short)0, a1a, false, false);
                a3a = __builtin_amdgcn_wmma_f32_16x16x32_bf16(false, A.v, false, B3a.v,
                                                              (short)0, a3a, false, false);
                a1b = __builtin_amdgcn_wmma_f32_16x16x32_bf16(false, A.v, false, B1b.v,
                                                              (short)0, a1b, false, false);
                a3b = __builtin_amdgcn_wmma_f32_16x16x32_bf16(false, A.v, false, B3b.v,
                                                              (short)0, a3b, false, false);
            }

            const bool vA = hcolA < HH;
            const bool vB = hcolB < HH;
#pragma unroll
            for (int v = 0; v < 8; ++v) {
                const int m = v + off8;
                const float sA = a1a[v] / (1.0f + __expf(-a1a[v]));
                const float sB = a1b[v] / (1.0f + __expf(-a1b[v]));
                Hs[m * HST + hcolA] = f2bf(vA ? (sA * a3a[v] * cw_s[m]) : 0.0f);
                Hs[m * HST + hcolB] = f2bf(vB ? (sB * a3b[v] * cw_s[m]) : 0.0f);
            }
        }
    } else {
        for (int ht = wave; ht < HPAD / 16; ht += 8) {
            const int h0   = ht * 16;
            const int hcol = h0 + nn;
            const int hrow = (hcol < HH) ? hcol : (HH - 1);
            const float* r1f = w1f + ((size_t)e * HH + hrow) * DD;
            const float* r3f = w3f + ((size_t)e * HH + hrow) * DD;

            v8f acc1 = {}; v8f acc3 = {};
            for (int ks = 0; ks < DD; ks += 32) {
                const int c0 = ks + khalf * 8;

                BF16x16 A;
                const unsigned short* ap = &Xs[nn * XST + c0];
                A.q[0] = *(const uint4*)ap;
                A.q[1] = *(const uint4*)(ap + 16);

                BF16x16 B1, B3;
                float4 f0 = *(const float4*)(r1f + c0);
                float4 f1 = *(const float4*)(r1f + c0 + 4);
                float4 f2 = *(const float4*)(r1f + c0 + 16);
                float4 f3 = *(const float4*)(r1f + c0 + 20);
                B1.u[0] = pk(f0.x, f0.y); B1.u[1] = pk(f0.z, f0.w);
                B1.u[2] = pk(f1.x, f1.y); B1.u[3] = pk(f1.z, f1.w);
                B1.u[4] = pk(f2.x, f2.y); B1.u[5] = pk(f2.z, f2.w);
                B1.u[6] = pk(f3.x, f3.y); B1.u[7] = pk(f3.z, f3.w);
                f0 = *(const float4*)(r3f + c0);
                f1 = *(const float4*)(r3f + c0 + 4);
                f2 = *(const float4*)(r3f + c0 + 16);
                f3 = *(const float4*)(r3f + c0 + 20);
                B3.u[0] = pk(f0.x, f0.y); B3.u[1] = pk(f0.z, f0.w);
                B3.u[2] = pk(f1.x, f1.y); B3.u[3] = pk(f1.z, f1.w);
                B3.u[4] = pk(f2.x, f2.y); B3.u[5] = pk(f2.z, f2.w);
                B3.u[6] = pk(f3.x, f3.y); B3.u[7] = pk(f3.z, f3.w);

                acc1 = __builtin_amdgcn_wmma_f32_16x16x32_bf16(false, A.v, false, B1.v,
                                                               (short)0, acc1, false, false);
                acc3 = __builtin_amdgcn_wmma_f32_16x16x32_bf16(false, A.v, false, B3.v,
                                                               (short)0, acc3, false, false);
            }

            const bool hvalid = hcol < HH;
#pragma unroll
            for (int v = 0; v < 8; ++v) {
                const int m = v + off8;
                const float s = acc1[v] / (1.0f + __expf(-acc1[v]));
                Hs[m * HST + hcol] = f2bf(hvalid ? (s * acc3[v] * cw_s[m]) : 0.0f);
            }
        }
    }
    __syncthreads();

    // ---- Phase 2: y = h W2^T (rows pre-weighted by cw), atomic combine ----
    if constexpr (PRE) {
        // two d-tiles per iteration: A fragment feeds 2 WMMAs
        for (int it = wave; it < DD / 32; it += 8) {
            const int n0a = it * 32;
            const int n0b = n0a + 16;
            const unsigned short* r2a = w2b + ((size_t)e * DD + n0a + nn) * HPAD;
            const unsigned short* r2c = w2b + ((size_t)e * DD + n0b + nn) * HPAD;

            v8f acca = {}, accb = {};
            for (int ks = 0; ks < HPAD; ks += 32) {
                const int c0 = ks + khalf * 8;

                BF16x16 A;
                const unsigned short* ap = &Hs[nn * HST + c0];
                A.q[0] = *(const uint4*)ap;
                A.q[1] = *(const uint4*)(ap + 16);

                BF16x16 Ba, Bb;
                Ba.q[0] = *(const uint4*)(r2a + c0);
                Ba.q[1] = *(const uint4*)(r2a + c0 + 16);
                Bb.q[0] = *(const uint4*)(r2c + c0);
                Bb.q[1] = *(const uint4*)(r2c + c0 + 16);

                acca = __builtin_amdgcn_wmma_f32_16x16x32_bf16(false, A.v, false, Ba.v,
                                                               (short)0, acca, false, false);
                accb = __builtin_amdgcn_wmma_f32_16x16x32_bf16(false, A.v, false, Bb.v,
                                                               (short)0, accb, false, false);
            }
#pragma unroll
            for (int v = 0; v < 8; ++v) {
                const int m = v + off8;
                const int tok = tok_s[m];
                if (tok >= 0) {
                    atomicAdd(&out[(size_t)tok * DD + n0a + nn], acca[v]);
                    atomicAdd(&out[(size_t)tok * DD + n0b + nn], accb[v]);
                }
            }
        }
    } else {
        for (int dt = wave; dt < DD / 16; dt += 8) {
            const int n0 = dt * 16;
            const float* r2f = w2f + ((size_t)e * DD + n0 + nn) * HH;

            v8f acc = {};
            for (int ks = 0; ks < HPAD; ks += 32) {
                const int c0 = ks + khalf * 8;

                BF16x16 A;
                const unsigned short* ap = &Hs[nn * HST + c0];
                A.q[0] = *(const uint4*)ap;
                A.q[1] = *(const uint4*)(ap + 16);

                BF16x16 B;
#pragma unroll
                for (int i = 0; i < 8; ++i) {
                    const int ca = c0 + i;
                    const int cb = c0 + 16 + i;
                    B.s[i]     = f2bf((ca < HH) ? r2f[ca] : 0.0f);
                    B.s[8 + i] = f2bf((cb < HH) ? r2f[cb] : 0.0f);
                }
                acc = __builtin_amdgcn_wmma_f32_16x16x32_bf16(false, A.v, false, B.v,
                                                              (short)0, acc, false, false);
            }
#pragma unroll
            for (int v = 0; v < 8; ++v) {
                const int m = v + off8;
                const int tok = tok_s[m];
                if (tok >= 0) atomicAdd(&out[(size_t)tok * DD + n0 + nn], acc[v]);
            }
        }
    }
}

// ---------------- host entry ----------------
extern "C" void kernel_launch(void* const* d_in, const int* in_sizes, int n_in,
                              void* d_out, int out_size, void* d_ws, size_t ws_size,
                              hipStream_t stream)
{
    const float* x  = (const float*)d_in[0];
    const float* wg = (const float*)d_in[1];
    const float* w1 = (const float*)d_in[2];
    const float* w2 = (const float*)d_in[3];
    const float* w3 = (const float*)d_in[4];
    float* out = (float*)d_out;

    char* ws = (char*)d_ws;
    int*   cnt     = (int*)ws;                                   // 8 ints
    float* probSum = (float*)(ws + 32);                          // 8 floats
    float* cw      = (float*)(ws + 64);                          // T*E floats
    int*   lists   = (int*)(ws + 64 + (size_t)TT * EE * sizeof(float)); // E*T ints

    const size_t w13n = (size_t)EE * HH * DD;                    // elems per w1/w3
    const size_t w2n  = (size_t)EE * DD * HPAD;                  // elems of padded w2
    size_t wbase = 64 + (size_t)TT * EE * 4 + (size_t)EE * TT * 4;
    wbase = (wbase + 255) & ~(size_t)255;
    unsigned short* w1b = (unsigned short*)(ws + wbase);
    unsigned short* w3b = w1b + w13n;
    unsigned short* w2b = w3b + w13n;
    const size_t need = wbase + (2 * w13n + w2n) * sizeof(unsigned short);
    const bool pre = (ws_size >= need);

    hipMemsetAsync(d_ws, 0, 64, stream);
    hipMemsetAsync(d_out, 0, (size_t)out_size * sizeof(float), stream);

    if (pre) {
        conv_copy_bf16<<<2048, 256, 0, stream>>>(w1, w1b, (long)(w13n / 4));
        conv_copy_bf16<<<2048, 256, 0, stream>>>(w3, w3b, (long)(w13n / 4));
        conv_w2_pad  <<<4096, 256, 0, stream>>>(w2, w2b);
    }

    gate_kernel<<<TT / 8, 256, 0, stream>>>(x, wg, cw, probSum, cnt, lists);
    loss_kernel<<<1, 32, 0, stream>>>(probSum, out + (size_t)TT * DD);

    if (pre)
        moe_main<true><<<EE * 256, 256, 0, stream>>>(x, w1, w2, w3, w1b, w2b, w3b,
                                                     cw, lists, cnt, out);
    else
        moe_main<false><<<EE * 256, 256, 0, stream>>>(x, w1, w2, w3, w1b, w2b, w3b,
                                                      cw, lists, cnt, out);
}
